// TopKSAE_919123001719
// MI455X (gfx1250) — matrix operand order repaired and checked by reading, not compiled, not run on hardware
//
#include <hip/hip_runtime.h>
#include <stdint.h>

#define D_IN   2048
#define D_SAE  32768
#define BATCH  8192
#define KTOP   64

#define TILE_M 128
#define TILE_N 128
#define TILE_K 32
#define LSTR   40   // padded LDS row stride in bf16 (80B: 16B-aligned, bank-spread)

typedef __attribute__((ext_vector_type(16))) __bf16 v16bf;
typedef __attribute__((ext_vector_type(8)))  __bf16 v8bf;
typedef __attribute__((ext_vector_type(8)))  float  v8f;

__device__ __forceinline__ v16bf make_frag(const __bf16* p0, const __bf16* p1) {
  v8bf a = *(const v8bf*)p0;
  v8bf b = *(const v8bf*)p1;
  return __builtin_shufflevector(a, b, 0,1,2,3,4,5,6,7,8,9,10,11,12,13,14,15);
}

__device__ __forceinline__ void split_store(__bf16* hi, __bf16* lo, float v) {
  __bf16 h = (__bf16)v;
  hi[0] = h;
  lo[0] = (__bf16)(v - (float)h);
}

// ---------------------------------------------------------------------------
// Kernel 1: pre = (x - b_dec) @ W_enc^T + b_enc   (written into z region)
// Double-buffered LDS, software-pipelined: next-tile global loads fly while
// current tile's 24 bf16x3 WMMAs execute; one barrier per K-step.
// ---------------------------------------------------------------------------
__global__ __launch_bounds__(256)
void sae_encode_gemm(const float* __restrict__ x, const float* __restrict__ W_enc,
                     const float* __restrict__ b_enc, const float* __restrict__ b_dec,
                     float* __restrict__ pre) {
  __shared__ __bf16 lAhi[2][TILE_M * LSTR];
  __shared__ __bf16 lAlo[2][TILE_M * LSTR];
  __shared__ __bf16 lBhi[2][TILE_N * LSTR];
  __shared__ __bf16 lBlo[2][TILE_N * LSTR];

  const int tid  = threadIdx.x;
  const int m0   = blockIdx.y * TILE_M;
  const int n0   = blockIdx.x * TILE_N;
  const int lane = tid & 31;
  const int wave = tid >> 5;
  const int wm   = (wave >> 1) * 32;   // 4 waves along M
  const int wn   = (wave &  1) * 64;   // 2 waves along N

  v8f acc[2][4] = {};

  const int lrow = tid >> 3;           // 0..31
  const int lcol = (tid & 7) * 4;      // 0..28 step 4

  float4 avr[4], bvr[4];

  auto load_tiles = [&](int k0) {
    float4 bd = *(const float4*)(b_dec + k0 + lcol);
#pragma unroll
    for (int j = 0; j < 4; ++j) {
      int r = lrow + j * 32;
      float4 av = *(const float4*)(x     + (size_t)(m0 + r) * D_IN + k0 + lcol);
      bvr[j]    = *(const float4*)(W_enc + (size_t)(n0 + r) * D_IN + k0 + lcol);
      av.x -= bd.x; av.y -= bd.y; av.z -= bd.z; av.w -= bd.w;
      avr[j] = av;
    }
  };

  auto store_tiles = [&](int buf) {
#pragma unroll
    for (int j = 0; j < 4; ++j) {
      int o = (lrow + j * 32) * LSTR + lcol;
      split_store(&lAhi[buf][o+0], &lAlo[buf][o+0], avr[j].x);
      split_store(&lAhi[buf][o+1], &lAlo[buf][o+1], avr[j].y);
      split_store(&lAhi[buf][o+2], &lAlo[buf][o+2], avr[j].z);
      split_store(&lAhi[buf][o+3], &lAlo[buf][o+3], avr[j].w);
      split_store(&lBhi[buf][o+0], &lBlo[buf][o+0], bvr[j].x);
      split_store(&lBhi[buf][o+1], &lBlo[buf][o+1], bvr[j].y);
      split_store(&lBhi[buf][o+2], &lBlo[buf][o+2], bvr[j].z);
      split_store(&lBhi[buf][o+3], &lBlo[buf][o+3], bvr[j].w);
    }
  };

  auto compute = [&](int buf) {
    // A fragments: 16-bit A 16x32 layout (K 0..7/16..23 lanes<16, 8..15/24..31 lanes>=16)
    const int krA = (lane < 16) ? 0 : 8;
    v16bf ah[2], al[2];
#pragma unroll
    for (int mi = 0; mi < 2; ++mi) {
      int row = wm + mi * 16 + (lane & 15);
      ah[mi] = make_frag(&lAhi[buf][row * LSTR + krA], &lAhi[buf][row * LSTR + krA + 16]);
      al[mi] = make_frag(&lAlo[buf][row * LSTR + krA], &lAlo[buf][row * LSTR + krA + 16]);
    }
    // B fragments: 32x16, lane=N, K 0..7/16..23 (lanes<16) / 8..15/24..31 (lanes>=16)
    const int krB = (lane < 16) ? 0 : 8;
#pragma unroll
    for (int ni = 0; ni < 4; ++ni) {
      int col = wn + ni * 16 + (lane & 15);
      v16bf bh = make_frag(&lBhi[buf][col * LSTR + krB], &lBhi[buf][col * LSTR + krB + 16]);
      v16bf bl = make_frag(&lBlo[buf][col * LSTR + krB], &lBlo[buf][col * LSTR + krB + 16]);
#pragma unroll
      for (int mi = 0; mi < 2; ++mi) {
        acc[mi][ni] = __builtin_amdgcn_wmma_f32_16x16x32_bf16(false, ah[mi], false, bh, (short)0, acc[mi][ni], false, false);
        acc[mi][ni] = __builtin_amdgcn_wmma_f32_16x16x32_bf16(false, ah[mi], false, bl, (short)0, acc[mi][ni], false, false);
        acc[mi][ni] = __builtin_amdgcn_wmma_f32_16x16x32_bf16(false, al[mi], false, bh, (short)0, acc[mi][ni], false, false);
      }
    }
  };

  const int nk = D_IN / TILE_K;
  load_tiles(0);
  store_tiles(0);
  __syncthreads();
  for (int kt = 0; kt < nk; ++kt) {
    if (kt + 1 < nk) load_tiles((kt + 1) * TILE_K);   // HBM loads in flight...
    compute(kt & 1);                                   // ...during WMMAs
    if (kt + 1 < nk) store_tiles((kt + 1) & 1);
    __syncthreads();
  }

  // epilogue: C/D 16x16 layout (VGPR r: M=r lanes<16, M=r+8 lanes>=16; N=lane%16)
  const int rofs  = (lane >> 4) * 8;
  const int nlane = lane & 15;
#pragma unroll
  for (int mi = 0; mi < 2; ++mi)
#pragma unroll
    for (int ni = 0; ni < 4; ++ni) {
      int gcol = n0 + wn + ni * 16 + nlane;
      float be = b_enc[gcol];
#pragma unroll
      for (int r = 0; r < 8; ++r) {
        int grow = m0 + wm + mi * 16 + rofs + r;
        pre[(size_t)grow * D_SAE + gcol] = acc[mi][ni][r] + be;   // b32 store (region 4B-offset)
      }
    }
}

// ---------------------------------------------------------------------------
// Kernel 2: per-row top-64 + relu, in place. Binary search on sortable keys.
// ---------------------------------------------------------------------------
__device__ __forceinline__ float key_to_float(uint32_t k) {
  uint32_t bits = (k & 0x80000000u) ? (k ^ 0x80000000u) : ~k;
  return __uint_as_float(bits);
}

__global__ __launch_bounds__(1024)
void sae_topk(float* __restrict__ z) {
  const int tid = threadIdx.x;
  float* row = z + (size_t)blockIdx.x * D_SAE;

  uint32_t key[32];
#pragma unroll
  for (int i = 0; i < 32; ++i) {
    uint32_t u = __float_as_uint(row[tid + i * 1024]);
    key[i] = u ^ ((u & 0x80000000u) ? 0xFFFFFFFFu : 0x80000000u);
  }

  __shared__ uint32_t wred[32];
  __shared__ uint32_t bc;
  __shared__ uint32_t scan[1024];

  // largest T with count(key >= T) >= KTOP
  uint32_t lo = 0u, hi = 0xFFFFFFFFu;
  for (int it = 0; it < 32; ++it) {
    uint32_t d = hi - lo;
    uint32_t mid = lo + (d >> 1) + (d & 1);
    int c = 0;
#pragma unroll
    for (int i = 0; i < 32; ++i) c += (key[i] >= mid) ? 1 : 0;
    for (int off = 16; off > 0; off >>= 1) c += __shfl_down(c, off, 32);
    if ((tid & 31) == 0) wred[tid >> 5] = (uint32_t)c;
    __syncthreads();
    if (tid == 0) { uint32_t s = 0; for (int w = 0; w < 32; ++w) s += wred[w]; bc = s; }
    __syncthreads();
    uint32_t total = bc;
    if (total >= KTOP) lo = mid; else hi = mid - 1;
    __syncthreads();
  }
  const uint32_t T = lo;

  // strictly-greater count -> number of tie slots remaining
  int cg = 0;
#pragma unroll
  for (int i = 0; i < 32; ++i) cg += (key[i] > T) ? 1 : 0;
  int cr = cg;
  for (int off = 16; off > 0; off >>= 1) cr += __shfl_down(cr, off, 32);
  if ((tid & 31) == 0) wred[tid >> 5] = (uint32_t)cr;
  __syncthreads();
  if (tid == 0) { uint32_t s = 0; for (int w = 0; w < 32; ++w) s += wred[w]; bc = s; }
  __syncthreads();
  const int rem = KTOP - (int)bc;

  // deterministic tie ranking: exclusive prefix over per-thread tie counts
  int tieCnt = 0;
#pragma unroll
  for (int i = 0; i < 32; ++i) tieCnt += (key[i] == T) ? 1 : 0;
  __syncthreads();
  scan[tid] = (uint32_t)tieCnt;
  __syncthreads();
  for (int off = 1; off < 1024; off <<= 1) {
    uint32_t t = (tid >= off) ? scan[tid - off] : 0u;
    __syncthreads();
    scan[tid] += t;
    __syncthreads();
  }
  const int myBase = (int)scan[tid] - tieCnt;

  int tie = 0;
#pragma unroll
  for (int i = 0; i < 32; ++i) {
    float outv = 0.0f;
    if (key[i] > T) {
      outv = fmaxf(key_to_float(key[i]), 0.0f);
    } else if (key[i] == T) {
      if (myBase + tie < rem) outv = fmaxf(key_to_float(key[i]), 0.0f);
      ++tie;
    }
    row[tid + i * 1024] = outv;
  }
}

// ---------------------------------------------------------------------------
// Kernel 3: x_hat = z @ W_dec^T + b_dec ; partial recon-loss per workgroup
// Same double-buffered pipeline; z region is 4B-offset -> scalar b32 loads.
// ---------------------------------------------------------------------------
__global__ __launch_bounds__(256)
void sae_decode_gemm(const float* __restrict__ z, const float* __restrict__ W_dec,
                     const float* __restrict__ b_dec, const float* __restrict__ x,
                     float* __restrict__ x_hat, float* __restrict__ ws) {
  __shared__ __bf16 lAhi[2][TILE_M * LSTR];
  __shared__ __bf16 lAlo[2][TILE_M * LSTR];
  __shared__ __bf16 lBhi[2][TILE_N * LSTR];
  __shared__ __bf16 lBlo[2][TILE_N * LSTR];
  __shared__ float  redbuf[256];

  const int tid  = threadIdx.x;
  const int m0   = blockIdx.y * TILE_M;
  const int n0   = blockIdx.x * TILE_N;
  const int lane = tid & 31;
  const int wave = tid >> 5;
  const int wm   = (wave >> 1) * 32;
  const int wn   = (wave &  1) * 64;

  v8f acc[2][4] = {};

  const int lrow = tid >> 3;
  const int lcol = (tid & 7) * 4;

  float4 avr[4], bvr[4];

  auto load_tiles = [&](int k0) {
#pragma unroll
    for (int j = 0; j < 4; ++j) {
      int r = lrow + j * 32;
      const float* zr = z + (size_t)(m0 + r) * D_SAE + k0 + lcol;
      avr[j].x = zr[0]; avr[j].y = zr[1]; avr[j].z = zr[2]; avr[j].w = zr[3];
      bvr[j]   = *(const float4*)(W_dec + (size_t)(n0 + r) * D_SAE + k0 + lcol);
    }
  };

  auto store_tiles = [&](int buf) {
#pragma unroll
    for (int j = 0; j < 4; ++j) {
      int o = (lrow + j * 32) * LSTR + lcol;
      split_store(&lAhi[buf][o+0], &lAlo[buf][o+0], avr[j].x);
      split_store(&lAhi[buf][o+1], &lAlo[buf][o+1], avr[j].y);
      split_store(&lAhi[buf][o+2], &lAlo[buf][o+2], avr[j].z);
      split_store(&lAhi[buf][o+3], &lAlo[buf][o+3], avr[j].w);
      split_store(&lBhi[buf][o+0], &lBlo[buf][o+0], bvr[j].x);
      split_store(&lBhi[buf][o+1], &lBlo[buf][o+1], bvr[j].y);
      split_store(&lBhi[buf][o+2], &lBlo[buf][o+2], bvr[j].z);
      split_store(&lBhi[buf][o+3], &lBlo[buf][o+3], bvr[j].w);
    }
  };

  auto compute = [&](int buf) {
    const int krA = (lane < 16) ? 0 : 8;
    v16bf ah[2], al[2];
#pragma unroll
    for (int mi = 0; mi < 2; ++mi) {
      int row = wm + mi * 16 + (lane & 15);
      ah[mi] = make_frag(&lAhi[buf][row * LSTR + krA], &lAhi[buf][row * LSTR + krA + 16]);
      al[mi] = make_frag(&lAlo[buf][row * LSTR + krA], &lAlo[buf][row * LSTR + krA + 16]);
    }
    const int krB = (lane < 16) ? 0 : 8;
#pragma unroll
    for (int ni = 0; ni < 4; ++ni) {
      int col = wn + ni * 16 + (lane & 15);
      v16bf bh = make_frag(&lBhi[buf][col * LSTR + krB], &lBhi[buf][col * LSTR + krB + 16]);
      v16bf bl = make_frag(&lBlo[buf][col * LSTR + krB], &lBlo[buf][col * LSTR + krB + 16]);
#pragma unroll
      for (int mi = 0; mi < 2; ++mi) {
        acc[mi][ni] = __builtin_amdgcn_wmma_f32_16x16x32_bf16(false, ah[mi], false, bh, (short)0, acc[mi][ni], false, false);
        acc[mi][ni] = __builtin_amdgcn_wmma_f32_16x16x32_bf16(false, ah[mi], false, bl, (short)0, acc[mi][ni], false, false);
        acc[mi][ni] = __builtin_amdgcn_wmma_f32_16x16x32_bf16(false, al[mi], false, bh, (short)0, acc[mi][ni], false, false);
      }
    }
  };

  const int nk = D_SAE / TILE_K;
  load_tiles(0);
  store_tiles(0);
  __syncthreads();
  for (int kt = 0; kt < nk; ++kt) {
    if (kt + 1 < nk) load_tiles((kt + 1) * TILE_K);
    compute(kt & 1);
    if (kt + 1 < nk) store_tiles((kt + 1) & 1);
    __syncthreads();
  }

  // epilogue: + b_dec, store x_hat, accumulate squared error
  const int rofs  = (lane >> 4) * 8;
  const int nlane = lane & 15;
  float lsum = 0.0f;
#pragma unroll
  for (int mi = 0; mi < 2; ++mi)
#pragma unroll
    for (int ni = 0; ni < 4; ++ni) {
      int gcol = n0 + wn + ni * 16 + nlane;
      float bd = b_dec[gcol];
#pragma unroll
      for (int r = 0; r < 8; ++r) {
        int grow = m0 + wm + mi * 16 + rofs + r;
        float xh = acc[mi][ni][r] + bd;
        x_hat[(size_t)grow * D_IN + gcol] = xh;     // b32 store (region 4B-offset)
        float d = xh - x[(size_t)grow * D_IN + gcol];
        lsum += d * d;
      }
    }

  __syncthreads();
  redbuf[tid] = lsum;
  __syncthreads();
  for (int s = 128; s > 0; s >>= 1) {
    if (tid < s) redbuf[tid] += redbuf[tid + s];
    __syncthreads();
  }
  if (tid == 0) ws[blockIdx.y * gridDim.x + blockIdx.x] = redbuf[0];
}

// ---------------------------------------------------------------------------
// Kernel 4: deterministic final loss reduction over 1024 partials
// ---------------------------------------------------------------------------
__global__ __launch_bounds__(256)
void sae_loss_reduce(const float* __restrict__ ws, float* __restrict__ loss) {
  __shared__ float buf[256];
  const int tid = threadIdx.x;
  float s = 0.0f;
#pragma unroll
  for (int j = 0; j < 4; ++j) s += ws[tid + j * 256];
  buf[tid] = s;
  __syncthreads();
  for (int st = 128; st > 0; st >>= 1) {
    if (tid < st) buf[tid] += buf[tid + st];
    __syncthreads();
  }
  if (tid == 0) loss[0] = buf[0] / (float)BATCH;
}

// ---------------------------------------------------------------------------
extern "C" void kernel_launch(void* const* d_in, const int* in_sizes, int n_in,
                              void* d_out, int out_size, void* d_ws, size_t ws_size,
                              hipStream_t stream) {
  const float* x     = (const float*)d_in[0];
  const float* W_enc = (const float*)d_in[1];
  const float* b_enc = (const float*)d_in[2];
  const float* W_dec = (const float*)d_in[3];
  const float* b_dec = (const float*)d_in[4];
  // d_in[5] = k (int scalar) : compile-time constant KTOP=64

  float* loss  = (float*)d_out;                       // output[0]
  float* x_hat = loss + 1;                            // output[1]: (B, D_IN)
  float* z     = x_hat + (size_t)BATCH * D_IN;        // output[2]: (B, D_SAE)
  float* ws    = (float*)d_ws;                        // 1024 partial loss sums

  dim3 g1(D_SAE / TILE_N, BATCH / TILE_M);            // 256 x 64
  sae_encode_gemm<<<g1, 256, 0, stream>>>(x, W_enc, b_enc, b_dec, z);

  sae_topk<<<BATCH, 1024, 0, stream>>>(z);

  dim3 g3(D_IN / TILE_N, BATCH / TILE_M);             // 16 x 64
  sae_decode_gemm<<<g3, 256, 0, stream>>>(z, W_dec, b_dec, x, x_hat, ws);

  sae_loss_reduce<<<1, 256, 0, stream>>>(ws, loss);
}
